// EncoderLayer_60120952209429
// MI455X (gfx1250) — matrix operand order repaired and checked by reading, not compiled, not run on hardware
//
#include <hip/hip_runtime.h>
#include <hip/hip_bf16.h>

typedef __attribute__((ext_vector_type(16))) _Float16 v16h;
typedef __attribute__((ext_vector_type(8)))  float    v8f;
typedef __attribute__((ext_vector_type(4)))  int      v4i;

#define BB   2
#define SS   2048
#define DD   768
#define HH   12
#define DKK  64
#define DFFF 3072
#define MM   (BB * SS)          // 4096 rows total

__device__ __forceinline__ v8f wmma_f16(v16h a, v16h b, v8f c) {
    // D = A(16x32 f16) * B(32x16 f16) + C(16x16 f32)
    return __builtin_amdgcn_wmma_f32_16x16x32_f16(
        false, a, false, b, (short)0, c, false, false);
}

__device__ __forceinline__ float gelu_exact(float v) {
    return 0.5f * v * (1.0f + erff(v * 0.70710678118654752440f));
}

// ---------------------------------------------------------------------------
// gfx1250 async global->LDS copies (16B), with portable fallback.
// Signature (from compiler diagnostic): (global int4*, local int4*, Ii, Ii)
// ---------------------------------------------------------------------------
#if __has_builtin(__builtin_amdgcn_global_load_async_to_lds_b128)
#define HAVE_ASYNC_LDS 1
typedef __attribute__((address_space(1))) v4i gv4i;
typedef __attribute__((address_space(3))) v4i lv4i;
__device__ __forceinline__ void async_cp16(const void* g, void* l) {
    __builtin_amdgcn_global_load_async_to_lds_b128((gv4i*)g, (lv4i*)l, 0, 0);
}
__device__ __forceinline__ void async_wait0() {
#if __has_builtin(__builtin_amdgcn_s_wait_asynccnt)
    __builtin_amdgcn_s_wait_asynccnt(0);
#else
    asm volatile("s_wait_asynccnt 0x0" ::: "memory");
#endif
}
#else
__device__ __forceinline__ void async_cp16(const void* g, void* l) {
    *(uint4*)l = *(const uint4*)g;
}
__device__ __forceinline__ void async_wait0() {}
#endif

// ---------------------------------------------------------------------------
// Elementwise f32 -> f16 conversion
// ---------------------------------------------------------------------------
__global__ __launch_bounds__(256) void cvt_f16_kernel(const float* __restrict__ src,
                                                      _Float16* __restrict__ dst, int n)
{
    const int i = (blockIdx.x * 256 + threadIdx.x) * 4;
    if (i + 3 < n) {
#pragma unroll
        for (int t = 0; t < 4; ++t) dst[i + t] = (_Float16)src[i + t];
    }
}

// ---------------------------------------------------------------------------
// Weight convert + transpose: Wt[n][k] = (f16) W[k][n].  64x64 tiles via LDS.
// grid (N/64, K/64), 256 threads.
// ---------------------------------------------------------------------------
__global__ __launch_bounds__(256) void wt_kernel(const float* __restrict__ W,
                                                 _Float16* __restrict__ Wt,
                                                 int K, int N)
{
    __shared__ __align__(16) _Float16 tile[64][72];
    const int t  = threadIdx.x;
    const int k0 = blockIdx.y * 64;
    const int n0 = blockIdx.x * 64;
    {
        const int r  = t >> 2;
        const int cs = (t & 3) * 16;
        const float* src = W + (size_t)(k0 + r) * N + n0 + cs;
#pragma unroll
        for (int i = 0; i < 16; ++i) tile[r][cs + i] = (_Float16)src[i];
    }
    __syncthreads();
    {
        const int n  = t >> 2;
        const int ks = (t & 3) * 16;
        union { uint4 u[2]; _Float16 h[16]; } tmp;
#pragma unroll
        for (int i = 0; i < 16; ++i) tmp.h[i] = tile[ks + i][n];
        uint4* dst = (uint4*)(Wt + (size_t)(n0 + n) * K + k0 + ks);
        dst[0] = tmp.u[0];
        dst[1] = tmp.u[1];
    }
}

// ---------------------------------------------------------------------------
// WMMA GEMM: C[M,N] = epi(A[M,K] @ Wt[N,K]^T + bias[N]),  A/Wt f16, bias f32.
// 128x128 tile, 8 waves (4x2), each wave 32x64.
// EPI: 0 = f32 store, 1 = f16 store, 2 = gelu->f16, 3 = f16 store to
//      V-transposed layout Vt[b][h][dk][s]  (requires Ndim == DD)
// ---------------------------------------------------------------------------
template <int EPI>
__global__ __launch_bounds__(256) void gemm_wmma_kernel(
    const _Float16* __restrict__ A, const _Float16* __restrict__ Wt,
    const float* __restrict__ bias, void* __restrict__ Cptr,
    int Kdim, int Ndim)
{
    __shared__ __align__(16) _Float16 a_lds[128][40];  // [m][k]
    __shared__ __align__(16) _Float16 b_lds[128][40];  // [n][k]

    const int tid  = threadIdx.x;
    const int lane = tid & 31;
    const int wid  = tid >> 5;
    const int wm   = wid >> 1;          // 0..3 -> row offset wm*32
    const int wn   = wid & 1;           // 0..1 -> col offset wn*64
    const int hid  = lane >> 4;
    const int lm   = lane & 15;

    const int rowBase = blockIdx.y * 128;
    const int colBase = blockIdx.x * 128;

    const int sr  = tid >> 1;           // 0..127 staging row
    const int seg = (tid & 1) * 16;     // 0 / 16

    v8f acc[2][4] = {};

    for (int k0 = 0; k0 < Kdim; k0 += 32) {
        const _Float16* ag = A  + (size_t)(rowBase + sr) * Kdim + k0 + seg;
        const _Float16* bg = Wt + (size_t)(colBase + sr) * Kdim + k0 + seg;
        if (k0 + 32 < Kdim) {           // prefetch next K tile
            __builtin_prefetch(ag + 32, 0, 0);
            __builtin_prefetch(bg + 32, 0, 0);
        }
        async_cp16(ag,     &a_lds[sr][seg]);
        async_cp16(ag + 8, &a_lds[sr][seg + 8]);
        async_cp16(bg,     &b_lds[sr][seg]);
        async_cp16(bg + 8, &b_lds[sr][seg + 8]);
        async_wait0();
        __syncthreads();

        v16h af[2];
#pragma unroll
        for (int i = 0; i < 2; ++i) {
            const int m = wm * 32 + i * 16 + lm;
#pragma unroll
            for (int t = 0; t < 8; ++t) af[i][t]     = a_lds[m][8 * hid + t];
#pragma unroll
            for (int t = 0; t < 8; ++t) af[i][8 + t] = a_lds[m][16 + 8 * hid + t];
        }
        v16h bf[4];
#pragma unroll
        for (int j = 0; j < 4; ++j) {
            const int n = wn * 64 + j * 16 + lm;
#pragma unroll
            for (int t = 0; t < 8; ++t) bf[j][t] = b_lds[n][8 * hid + t];
#pragma unroll
            for (int t = 0; t < 8; ++t) bf[j][8 + t] = b_lds[n][16 + 8 * hid + t];
        }
#pragma unroll
        for (int i = 0; i < 2; ++i)
#pragma unroll
            for (int j = 0; j < 4; ++j)
                acc[i][j] = wmma_f16(af[i], bf[j], acc[i][j]);

        __syncthreads();
    }

    // ---- epilogue ----
#pragma unroll
    for (int i = 0; i < 2; ++i) {
#pragma unroll
        for (int j = 0; j < 4; ++j) {
            const int ncol = colBase + wn * 64 + j * 16 + lm;
            const float bv = bias[ncol];
            if (EPI == 3) {
                // store 8 consecutive s-positions as one 16B chunk
                const int h  = ncol >> 6, dk = ncol & 63;
                const int m0 = rowBase + wm * 32 + i * 16 + 8 * hid;
                const int b  = m0 >> 11, s0 = m0 & (SS - 1);
                union { uint4 u; _Float16 hh[8]; } tmp;
#pragma unroll
                for (int r = 0; r < 8; ++r)
                    tmp.hh[r] = (_Float16)(acc[i][j][r] + bv);
                *(uint4*)((_Float16*)Cptr +
                          (((size_t)(b * HH + h) * 64 + dk) * SS + s0)) = tmp.u;
            } else {
#pragma unroll
                for (int r = 0; r < 8; ++r) {
                    const int mrow = rowBase + wm * 32 + i * 16 + r + 8 * hid;
                    float v = acc[i][j][r] + bv;
                    if (EPI == 2) v = gelu_exact(v);
                    const size_t off = (size_t)mrow * Ndim + ncol;
                    if (EPI == 0) ((float*)Cptr)[off]    = v;
                    else          ((_Float16*)Cptr)[off] = (_Float16)v;
                }
            }
        }
    }
}

// ---------------------------------------------------------------------------
// Per-64-chunk L2 row-normalization in place (f16), one wave per chunk
// ---------------------------------------------------------------------------
__global__ __launch_bounds__(256) void rownorm_kernel(_Float16* __restrict__ X, int nchunks)
{
    const int gw   = (blockIdx.x * blockDim.x + threadIdx.x) >> 5;
    const int lane = threadIdx.x & 31;
    if (gw >= nchunks) return;
    const size_t base = (size_t)gw * 64;
    float a = (float)X[base + lane];
    float b = (float)X[base + 32 + lane];
    float ss = a * a + b * b;
#pragma unroll
    for (int off = 16; off > 0; off >>= 1) ss += __shfl_xor(ss, off, 32);
    const float sc = 1.0f / fmaxf(sqrtf(ss), 1e-12f);
    X[base + lane]      = (_Float16)(a * sc);
    X[base + 32 + lane] = (_Float16)(b * sc);
}

// ---------------------------------------------------------------------------
// Fused squared attention: O = (Qn Kn^T)^2 @ V, streamed over kv.
// Qn/Kn: [M][D] f16 ; Vt: [b][h][dk][S] f16 ; O: [M][D] f16.
// Grid (S/128, H, B), 256 threads = 8 waves, each wave 16 q rows.
// ---------------------------------------------------------------------------
__global__ __launch_bounds__(256) void attn_kernel(
    const _Float16* __restrict__ Qn, const _Float16* __restrict__ Kn,
    const _Float16* __restrict__ Vt, _Float16* __restrict__ O)
{
    __shared__ __align__(16) _Float16 kn_lds[32][72];       // [kv_local][dk]
    __shared__ __align__(16) _Float16 vt_lds[64][40];       // [dk][kv_local]
    __shared__ __align__(16) _Float16 att_lds[8][16][40];   // per wave [m][kv_local]

    const int tid  = threadIdx.x;
    const int lane = tid & 31;
    const int wid  = tid >> 5;
    const int hid  = lane >> 4;
    const int lm   = lane & 15;

    const int b  = blockIdx.z;
    const int h  = blockIdx.y;
    const int qb = blockIdx.x;
    const int hc = h * DKK;
    const int bh = b * HH + h;

    // Q fragments: 16 rows x 64 (two K=32 chunks) straight from global
    const size_t qrow = (size_t)b * SS + qb * 128 + wid * 16 + lm;
    v16h aq[2];
#pragma unroll
    for (int c = 0; c < 2; ++c) {
        const _Float16* src = Qn + qrow * DD + hc + c * 32;
#pragma unroll
        for (int t = 0; t < 8; ++t) aq[c][t]     = src[8 * hid + t];
#pragma unroll
        for (int t = 0; t < 8; ++t) aq[c][8 + t] = src[16 + 8 * hid + t];
    }

    v8f oacc[4] = {};

    for (int kv = 0; kv < SS; kv += 32) {
        // cooperative stage: Kn chunk 32x64, Vt chunk 64x32 (all 16B copies)
        {
            const int kr  = tid >> 3;
            const int col = (tid & 7) * 8;
            async_cp16(Kn + ((size_t)b * SS + kv + kr) * DD + hc + col,
                       &kn_lds[kr][col]);
            const int dr  = tid >> 2;
            const int ks  = (tid & 3) * 8;
            async_cp16(Vt + ((size_t)bh * 64 + dr) * SS + kv + ks,
                       &vt_lds[dr][ks]);
        }
        async_wait0();
        __syncthreads();

        // scores (two 16x16 kv subtiles), square -> f16 A-fragment via LDS
#pragma unroll
        for (int nt = 0; nt < 2; ++nt) {
            v8f sc = {};
#pragma unroll
            for (int c = 0; c < 2; ++c) {
                v16h bk;
                const _Float16* p = &kn_lds[nt * 16 + lm][c * 32];
#pragma unroll
                for (int t = 0; t < 8; ++t) bk[t] = p[8 * hid + t];
#pragma unroll
                for (int t = 0; t < 8; ++t) bk[8 + t] = p[16 + 8 * hid + t];
                sc = wmma_f16(aq[c], bk, sc);
            }
#pragma unroll
            for (int r = 0; r < 8; ++r) {
                const float sq = sc[r] * sc[r];
                att_lds[wid][r + 8 * hid][nt * 16 + lm] = (_Float16)sq;
            }
        }

        // reload attn chunk in A-fragment layout (same-wave LDS, in-order)
        v16h aa;
#pragma unroll
        for (int t = 0; t < 8; ++t) aa[t]     = att_lds[wid][lm][8 * hid + t];
#pragma unroll
        for (int t = 0; t < 8; ++t) aa[8 + t] = att_lds[wid][lm][16 + 8 * hid + t];

        // O += attn(16x32) @ V(32x64)  -- V fragments in table k order
#pragma unroll
        for (int j = 0; j < 4; ++j) {
            v16h bv;
            const _Float16* p = &vt_lds[j * 16 + lm][0];
#pragma unroll
            for (int t = 0; t < 8; ++t) bv[t] = p[8 * hid + t];
#pragma unroll
            for (int t = 0; t < 8; ++t) bv[8 + t] = p[16 + 8 * hid + t];
            oacc[j] = wmma_f16(aa, bv, oacc[j]);
        }
        __syncthreads();
    }

#pragma unroll
    for (int j = 0; j < 4; ++j)
#pragma unroll
        for (int r = 0; r < 8; ++r) {
            const size_t row = (size_t)b * SS + qb * 128 + wid * 16 + r + 8 * hid;
            O[row * DD + hc + j * 16 + lm] = (_Float16)oacc[j][r];
        }
}

// ---------------------------------------------------------------------------
// out[row] = LayerNorm(x[row] + y[row]) * gamma + beta ; optional f16 copy
// ---------------------------------------------------------------------------
__global__ __launch_bounds__(256) void add_ln_kernel(
    const float* __restrict__ x, const float* __restrict__ y,
    const float* __restrict__ gamma, const float* __restrict__ beta,
    float* __restrict__ out, _Float16* __restrict__ out_h)
{
    const int row = blockIdx.x;
    const int tid = threadIdx.x;
    const int wid = tid >> 5, lane = tid & 31;
    __shared__ float rs[8], rss[8];

    float v[3], s = 0.f, ss = 0.f;
#pragma unroll
    for (int i = 0; i < 3; ++i) {
        const int c = tid + i * 256;
        const float t = x[(size_t)row * DD + c] + y[(size_t)row * DD + c];
        v[i] = t; s += t; ss += t * t;
    }
#pragma unroll
    for (int off = 16; off > 0; off >>= 1) {
        s += __shfl_xor(s, off, 32); ss += __shfl_xor(ss, off, 32);
    }
    if (lane == 0) { rs[wid] = s; rss[wid] = ss; }
    __syncthreads();
    if (wid == 0) {
        s  = (lane < 8) ? rs[lane]  : 0.f;
        ss = (lane < 8) ? rss[lane] : 0.f;
#pragma unroll
        for (int off = 4; off > 0; off >>= 1) {
            s += __shfl_xor(s, off, 32); ss += __shfl_xor(ss, off, 32);
        }
        if (lane == 0) { rs[0] = s; rss[0] = ss; }
    }
    __syncthreads();
    const float mu  = rs[0] * (1.0f / DD);
    const float var = rss[0] * (1.0f / DD) - mu * mu;
    const float inv = rsqrtf(var + 1e-5f);
#pragma unroll
    for (int i = 0; i < 3; ++i) {
        const int c = tid + i * 256;
        const float r = (v[i] - mu) * inv * gamma[c] + beta[c];
        out[(size_t)row * DD + c] = r;
        if (out_h) out_h[(size_t)row * DD + c] = (_Float16)r;
    }
}

// ---------------------------------------------------------------------------
extern "C" void kernel_launch(void* const* d_in, const int* in_sizes, int n_in,
                              void* d_out, int out_size, void* d_ws, size_t ws_size,
                              hipStream_t stream)
{
    const float* x     = (const float*)d_in[0];
    const float* Wq    = (const float*)d_in[1];
    const float* bq    = (const float*)d_in[2];
    const float* Wk    = (const float*)d_in[3];
    const float* bk    = (const float*)d_in[4];
    const float* Wv    = (const float*)d_in[5];
    const float* bv    = (const float*)d_in[6];
    const float* Wo    = (const float*)d_in[7];
    const float* bo    = (const float*)d_in[8];
    const float* W1    = (const float*)d_in[9];
    const float* b1    = (const float*)d_in[10];
    const float* W2    = (const float*)d_in[11];
    const float* b2    = (const float*)d_in[12];
    const float* gamma = (const float*)d_in[13];
    const float* beta  = (const float*)d_in[14];
    float* out = (float*)d_out;

    // workspace layout (all offsets multiple of 256B)
    char* w = (char*)d_ws;
    const size_t szMD_h   = (size_t)MM * DD * sizeof(_Float16);    // 6.29 MB
    const size_t szMD_f   = (size_t)MM * DD * sizeof(float);       // 12.58 MB
    const size_t szMDFF_h = (size_t)MM * DFFF * sizeof(_Float16);  // 25.17 MB
    const size_t szW_h    = (size_t)DD * DD * sizeof(_Float16);    // 1.18 MB
    const size_t szW1_h   = (size_t)DD * DFFF * sizeof(_Float16);  // 4.72 MB

    size_t o = 0;
    _Float16* xh  = (_Float16*)(w + o); o += szMD_h;
    _Float16* qn  = (_Float16*)(w + o); o += szMD_h;
    _Float16* kn  = (_Float16*)(w + o); o += szMD_h;
    _Float16* vt  = (_Float16*)(w + o); o += szMD_h;   // [b][h][dk][S]
    _Float16* oh  = (_Float16*)(w + o); o += szMD_h;
    _Float16* x1h = (_Float16*)(w + o); o += szMD_h;
    float*    x1  = (float*)   (w + o); o += szMD_f;
    float*    ff  = (float*)   (w + o); o += szMD_f;
    _Float16* hb  = (_Float16*)(w + o); o += szMDFF_h;
    _Float16* wqt = (_Float16*)(w + o); o += szW_h;
    _Float16* wkt = (_Float16*)(w + o); o += szW_h;
    _Float16* wvt = (_Float16*)(w + o); o += szW_h;
    _Float16* wot = (_Float16*)(w + o); o += szW_h;
    _Float16* w1t = (_Float16*)(w + o); o += szW1_h;
    _Float16* w2t = (_Float16*)(w + o); o += szW1_h;

    const dim3 blk(256);
    const dim3 gD(DD / 128, MM / 128);      // (6, 32)
    const dim3 gFF(DFFF / 128, MM / 128);   // (24, 32)

    // activations -> f16 ; weights -> f16 transposed [N][K]
    cvt_f16_kernel<<<(MM * DD) / 1024, blk, 0, stream>>>(x, xh, MM * DD);
    wt_kernel<<<dim3(DD / 64, DD / 64),   blk, 0, stream>>>(Wq, wqt, DD, DD);
    wt_kernel<<<dim3(DD / 64, DD / 64),   blk, 0, stream>>>(Wk, wkt, DD, DD);
    wt_kernel<<<dim3(DD / 64, DD / 64),   blk, 0, stream>>>(Wv, wvt, DD, DD);
    wt_kernel<<<dim3(DD / 64, DD / 64),   blk, 0, stream>>>(Wo, wot, DD, DD);
    wt_kernel<<<dim3(DFFF / 64, DD / 64), blk, 0, stream>>>(W1, w1t, DD, DFFF);
    wt_kernel<<<dim3(DD / 64, DFFF / 64), blk, 0, stream>>>(W2, w2t, DFFF, DD);

    // QKV projections (V stored transposed per head)
    gemm_wmma_kernel<1><<<gD, blk, 0, stream>>>(xh, wqt, bq, qn, DD, DD);
    gemm_wmma_kernel<1><<<gD, blk, 0, stream>>>(xh, wkt, bk, kn, DD, DD);
    gemm_wmma_kernel<3><<<gD, blk, 0, stream>>>(xh, wvt, bv, vt, DD, DD);

    // per-head L2 row-norm of Q, K
    const int nchunks = MM * HH;  // 49152
    rownorm_kernel<<<(nchunks * 32) / 256, blk, 0, stream>>>(qn, nchunks);
    rownorm_kernel<<<(nchunks * 32) / 256, blk, 0, stream>>>(kn, nchunks);

    // fused squared attention
    attn_kernel<<<dim3(SS / 128, HH, BB), blk, 0, stream>>>(qn, kn, vt, oh);

    // output projection + first residual LN (also emit f16 copy of x1)
    gemm_wmma_kernel<0><<<gD, blk, 0, stream>>>(oh, wot, bo, ff, DD, DD);
    add_ln_kernel<<<MM, blk, 0, stream>>>(x, ff, gamma, beta, x1, x1h);

    // FFN
    gemm_wmma_kernel<2><<<gFF, blk, 0, stream>>>(x1h, w1t, b1, hb, DD, DFFF);
    gemm_wmma_kernel<0><<<gD,  blk, 0, stream>>>(hb, w2t, b2, ff, DFFF, DD);
    add_ln_kernel<<<MM, blk, 0, stream>>>(x1, ff, gamma, beta, out, nullptr);
}
